// CombinedSAGEGAT_78580721647967
// MI455X (gfx1250) — matrix operand order repaired and checked by reading, not compiled, not run on hardware
//
#include <hip/hip_runtime.h>
#include <hip/hip_bf16.h>
#include <cstddef>

typedef __attribute__((ext_vector_type(16))) _Float16 v16h;
typedef __attribute__((ext_vector_type(8)))  float    v8f;

#define LANE (threadIdx.x & 31)

// ---------------------------------------------------------------------------
// WMMA helpers (CDNA5 wave32, v_wmma_f32_16x16x32_f16)
// A: 16x32 f16  lane<16: M=lane, halves = K{0..7},{16..23}; lane>=16: M=lane-16, K{8..15},{24..31}
// B: 32x16 f16  lane: N=lane&15, halves t: K = (lane<16?0:8)+t+(t&8)
// C/D: 16x16 f32 lane: N=lane&15, reg r: M = r + (lane<16?0:8)
// ---------------------------------------------------------------------------
__device__ __forceinline__ v8f wmma16(v16h a, v16h b, v8f c) {
  return __builtin_amdgcn_wmma_f32_16x16x32_f16(false, a, false, b, (short)0, c, false, false);
}

// Branch-free fast activations (v_exp_f32 + v_rcp_f32; no libm slow paths,
// keeps EXEC untouched around the WMMA region).
__device__ __forceinline__ float sigm(float x) {
  return __builtin_amdgcn_rcpf(1.0f + __expf(-x));
}
__device__ __forceinline__ float tanh_fast(float x) {
  return 1.0f - 2.0f * __builtin_amdgcn_rcpf(__expf(2.0f * x) + 1.0f);
}

// A-operand fragment fully from a global f32 row (ld >= 32, 16B aligned):
// halves 0..7 = row[kb..kb+7], 8..15 = row[16+kb..16+kb+7]
__device__ __forceinline__ v16h make_a_g(const float* row, int kb) {
  const float4* p = (const float4*)(row + kb);
  const float4* q = (const float4*)(row + 16 + kb);
  float4 a0 = p[0], a1 = p[1], b0 = q[0], b1 = q[1];
  v16h a;
  a[0] = (_Float16)a0.x;  a[1] = (_Float16)a0.y;  a[2] = (_Float16)a0.z;  a[3] = (_Float16)a0.w;
  a[4] = (_Float16)a1.x;  a[5] = (_Float16)a1.y;  a[6] = (_Float16)a1.z;  a[7] = (_Float16)a1.w;
  a[8] = (_Float16)b0.x;  a[9] = (_Float16)b0.y;  a[10] = (_Float16)b0.z; a[11] = (_Float16)b0.w;
  a[12] = (_Float16)b1.x; a[13] = (_Float16)b1.y; a[14] = (_Float16)b1.z; a[15] = (_Float16)b1.w;
  return a;
}

// A = [X(16 global f32 cols) | H(16 LDS f16 cols)] K-concat
__device__ __forceinline__ v16h make_a_x8_h8(const float* xrow, int kb, const _Float16* hrow) {
  const float4* p = (const float4*)(xrow + kb);
  float4 a0 = p[0], a1 = p[1];
  v16h a;
  a[0] = (_Float16)a0.x; a[1] = (_Float16)a0.y; a[2] = (_Float16)a0.z; a[3] = (_Float16)a0.w;
  a[4] = (_Float16)a1.x; a[5] = (_Float16)a1.y; a[6] = (_Float16)a1.z; a[7] = (_Float16)a1.w;
#pragma unroll
  for (int j = 0; j < 8; ++j) a[8 + j] = hrow[kb + j];
  return a;
}

// A-operand from LDS f16 tile (row-major, row stride ld, K-chunk of 32)
__device__ __forceinline__ v16h load_a_lds(const _Float16* A, int ld) {
  const int m = LANE & 15, kb = (LANE < 16) ? 0 : 8;
  v16h a;
#pragma unroll
  for (int j = 0; j < 8; ++j) a[j] = A[m * ld + kb + j];
#pragma unroll
  for (int j = 0; j < 8; ++j) a[8 + j] = A[m * ld + 16 + kb + j];
  return a;
}

__device__ __forceinline__ v16h load_b_lds(const _Float16* B, int ld, int col0) {
  const int nn = (LANE & 15) + col0, kb = (LANE < 16) ? 0 : 8;
  v16h b;
#pragma unroll
  for (int t = 0; t < 16; ++t) b[t] = B[(kb + t + (t & 8)) * ld + nn];
  return b;
}

__device__ __forceinline__ v16h load_b_g32(const float* W, int ld, int col0) {
  const int nn = (LANE & 15) + col0, kb = (LANE < 16) ? 0 : 8;
  v16h b;
#pragma unroll
  for (int t = 0; t < 16; ++t) b[t] = (_Float16)W[(size_t)(kb + t + (t & 8)) * ld + nn];
  return b;
}

// ---------------------------------------------------------------------------
// Layer 1: SAGE-LSTM, F=16, hidden=16, out=32.  One wave per 16 nodes.
// Each LSTM step: [x_t | h](16x32) @ [wih.T ; whh.T](32x64) -> 4 gate tiles.
// X half of A comes straight from global (b128 loads); only h round-trips LDS.
// ---------------------------------------------------------------------------
__global__ __launch_bounds__(128)
void sage1_kernel(const float* __restrict__ x, const long long* __restrict__ nbr,
                  const float* __restrict__ wih, const float* __restrict__ whh,
                  const float* __restrict__ bih, const float* __restrict__ bhh,
                  const float* __restrict__ wself, const float* __restrict__ wneigh,
                  const float* __restrict__ bout, float* __restrict__ out, int N) {
  __shared__ _Float16 sB[32][64];      // [wih.T ; whh.T]
  __shared__ _Float16 sW[32][32];      // [wself ; wneigh]
  __shared__ _Float16 sH[4][16][16];
  __shared__ int      sIdx[4][16][16];

  const int tid = threadIdx.x, lane = tid & 31, wv = tid >> 5;
  const int wavesTotal = (N + 15) >> 4;
  const int wave = blockIdx.x * 4 + wv;
  const bool active = wave < wavesTotal;
  const int nb = (active ? wave : 0) * 16;

  for (int e = tid; e < 32 * 64; e += 128) {
    int k = e >> 6, g = e & 63;
    sB[k][g] = (_Float16)((k < 16) ? wih[g * 16 + k] : whh[g * 16 + (k - 16)]);
  }
  for (int e = tid; e < 32 * 32; e += 128) {
    int k = e >> 5, j = e & 31;
    sW[k][j] = (_Float16)((k < 16) ? wself[k * 32 + j] : wneigh[(k - 16) * 32 + j]);
  }
  for (int e = lane; e < 256; e += 32) {
    int m = e >> 4, d = e & 15;
    sIdx[wv][m][d] = (int)nbr[(size_t)(nb + m) * 16 + d];
    sH[wv][m][d] = (_Float16)0.f;
  }

  const int n = lane & 15, m = lane & 15;
  const int kb = (lane < 16) ? 0 : 8;
  const int rowoff = (lane < 16) ? 0 : 8;
  const float bI = bih[n] + bhh[n], bF = bih[16 + n] + bhh[16 + n];
  const float bG = bih[32 + n] + bhh[32 + n], bO = bih[48 + n] + bhh[48 + n];

  v8f c = {};
  for (int t = 0; t < 16; ++t) {
    __syncthreads();                    // h (and idx/weights) from prev iter visible
    int src = sIdx[wv][m][t];
    v16h a = make_a_x8_h8(x + (size_t)src * 16, kb, &sH[wv][m][0]);
    v8f z = {};
    v8f gi = wmma16(a, load_b_lds(&sB[0][0], 64, 0), z);
    v8f gf = wmma16(a, load_b_lds(&sB[0][0], 64, 16), z);
    v8f gg = wmma16(a, load_b_lds(&sB[0][0], 64, 32), z);
    v8f go = wmma16(a, load_b_lds(&sB[0][0], 64, 48), z);
    __syncthreads();                    // all lanes done reading sH
#pragma unroll
    for (int r = 0; r < 8; ++r) {
      float cv = sigm(gf[r] + bF) * c[r] + sigm(gi[r] + bI) * tanh_fast(gg[r] + bG);
      c[r] = cv;
      sH[wv][r + rowoff][n] = (_Float16)(sigm(go[r] + bO) * tanh_fast(cv));
    }
  }
  // out = relu([x_self | h] @ [wself ; wneigh] + b)
  __syncthreads();
  v16h a2 = make_a_x8_h8(x + (size_t)(nb + m) * 16, kb, &sH[wv][m][0]);
  v8f z = {};
  v8f o0 = wmma16(a2, load_b_lds(&sW[0][0], 32, 0), z);
  v8f o1 = wmma16(a2, load_b_lds(&sW[0][0], 32, 16), z);
  if (active) {
#pragma unroll
    for (int r = 0; r < 8; ++r) {
      int mm = nb + r + rowoff;
      float v0 = o0[r] + bout[n];      v0 = v0 > 0.f ? v0 : 0.f;
      float v1 = o1[r] + bout[16 + n]; v1 = v1 > 0.f ? v1 : 0.f;
      out[(size_t)mm * 32 + n] = v0;
      out[(size_t)mm * 32 + 16 + n] = v1;
    }
  }
}

// ---------------------------------------------------------------------------
// Layer 2: SAGE-LSTM, F=32, hidden=32.  Step: [x_t|h](16x64) @ (64x128) -> 8 tiles.
// ---------------------------------------------------------------------------
__global__ __launch_bounds__(128)
void sage2_kernel(const float* __restrict__ hin, const long long* __restrict__ nbr,
                  const float* __restrict__ wih, const float* __restrict__ whh,
                  const float* __restrict__ bih, const float* __restrict__ bhh,
                  const float* __restrict__ wself, const float* __restrict__ wneigh,
                  const float* __restrict__ bout, float* __restrict__ out, int N) {
  __shared__ _Float16 sB[64][128];
  __shared__ _Float16 sW[64][32];
  __shared__ _Float16 sH[4][16][32];
  __shared__ int      sIdx[4][16][16];

  const int tid = threadIdx.x, lane = tid & 31, wv = tid >> 5;
  const int wavesTotal = (N + 15) >> 4;
  const int wave = blockIdx.x * 4 + wv;
  const bool active = wave < wavesTotal;
  const int nb = (active ? wave : 0) * 16;

  for (int e = tid; e < 64 * 128; e += 128) {
    int k = e >> 7, g = e & 127;
    sB[k][g] = (_Float16)((k < 32) ? wih[g * 32 + k] : whh[g * 32 + (k - 32)]);
  }
  for (int e = tid; e < 64 * 32; e += 128) {
    int k = e >> 5, j = e & 31;
    sW[k][j] = (_Float16)((k < 32) ? wself[k * 32 + j] : wneigh[(k - 32) * 32 + j]);
  }
  for (int e = lane; e < 256; e += 32)
    sIdx[wv][e >> 4][e & 15] = (int)nbr[(size_t)(nb + (e >> 4)) * 16 + (e & 15)];
  for (int e = lane; e < 512; e += 32) sH[wv][e >> 5][e & 31] = (_Float16)0.f;

  const int n = lane & 15, m = lane & 15;
  const int kb = (lane < 16) ? 0 : 8;
  const int rowoff = (lane < 16) ? 0 : 8;
  float bs[8];
#pragma unroll
  for (int t = 0; t < 8; ++t) bs[t] = bih[t * 16 + n] + bhh[t * 16 + n];

  v8f c0 = {}, c1 = {};
  for (int t = 0; t < 16; ++t) {
    __syncthreads();
    int src = sIdx[wv][m][t];
    v16h a0 = make_a_g(hin + (size_t)src * 32, kb);     // x_t chunk (K 0..31)
    v16h a1 = load_a_lds(&sH[wv][0][0], 32);            // h chunk   (K 32..63)
    v8f g[8];
#pragma unroll
    for (int tt = 0; tt < 8; ++tt) {
      v8f z = {};
      g[tt] = wmma16(a0, load_b_lds(&sB[0][0], 128, tt * 16), z);
      g[tt] = wmma16(a1, load_b_lds(&sB[32][0], 128, tt * 16), g[tt]);
    }
    __syncthreads();
#pragma unroll
    for (int p = 0; p < 2; ++p) {
      v8f* cp = p ? &c1 : &c0;
#pragma unroll
      for (int r = 0; r < 8; ++r) {
        float cv = sigm(g[2 + p][r] + bs[2 + p]) * (*cp)[r] +
                   sigm(g[p][r] + bs[p]) * tanh_fast(g[4 + p][r] + bs[4 + p]);
        (*cp)[r] = cv;
        sH[wv][r + rowoff][p * 16 + n] = (_Float16)(sigm(g[6 + p][r] + bs[6 + p]) * tanh_fast(cv));
      }
    }
  }
  // out = relu([h_self | h_lstm] @ [wself ; wneigh] + b)
  __syncthreads();
  v16h a0 = make_a_g(hin + (size_t)(nb + m) * 32, kb);
  v16h a1 = load_a_lds(&sH[wv][0][0], 32);
  v8f z = {};
  v8f o0 = wmma16(a0, load_b_lds(&sW[0][0], 32, 0), z);
  o0 = wmma16(a1, load_b_lds(&sW[32][0], 32, 0), o0);
  v8f o1 = wmma16(a0, load_b_lds(&sW[0][0], 32, 16), z);
  o1 = wmma16(a1, load_b_lds(&sW[32][0], 32, 16), o1);
  if (active) {
#pragma unroll
    for (int r = 0; r < 8; ++r) {
      int mm = nb + r + rowoff;
      float v0 = o0[r] + bout[n];      v0 = v0 > 0.f ? v0 : 0.f;
      float v1 = o1[r] + bout[16 + n]; v1 = v1 > 0.f ? v1 : 0.f;
      out[(size_t)mm * 32 + n] = v0;
      out[(size_t)mm * 32 + 16 + n] = v1;
    }
  }
}

// ---------------------------------------------------------------------------
// GAT projection GEMM: C[N,128] = A[N,K] @ W[K,128].  One wave per 16 rows.
// ---------------------------------------------------------------------------
__global__ __launch_bounds__(128)
void proj_kernel(const float* __restrict__ A, const float* __restrict__ W,
                 float* __restrict__ C, int N, int K) {
  const int lane = threadIdx.x & 31, wv = threadIdx.x >> 5;
  const long wave = (long)blockIdx.x * 4 + wv;
  const long r0 = wave * 16;
  if (r0 >= N) return;
  const int m = lane & 15, kb = (lane < 16) ? 0 : 8;
  v8f acc[8];
  v8f z = {};
#pragma unroll
  for (int t = 0; t < 8; ++t) acc[t] = z;
  for (int kc = 0; kc < K; kc += 32) {
    v16h a = make_a_g(A + (size_t)(r0 + m) * K + kc, kb);
#pragma unroll
    for (int t = 0; t < 8; ++t) {
      v16h b = load_b_g32(W + (size_t)kc * 128, 128, t * 16);
      acc[t] = wmma16(a, b, acc[t]);
    }
  }
  const int n = lane & 15, rowoff = (lane < 16) ? 0 : 8;
#pragma unroll
  for (int t = 0; t < 8; ++t)
#pragma unroll
    for (int r = 0; r < 8; ++r)
      C[(size_t)(r0 + r + rowoff) * 128 + t * 16 + n] = acc[t][r];
}

// ---------------------------------------------------------------------------
// GATv2 edge softmax + aggregation. One wave per node; lane = output channel.
// mode 0/1: elu -> out[N,128]; mode 2: mean over heads -> out[N,32].
// ---------------------------------------------------------------------------
__global__ __launch_bounds__(256)
void gat_edge_kernel(const float* __restrict__ fs, const float* __restrict__ fd,
                     const long long* __restrict__ nbr,
                     const float* __restrict__ attn, const float* __restrict__ bias,
                     float* __restrict__ out, int mode, int N) {
  const int lane = threadIdx.x & 31, wv = threadIdx.x >> 5;
  const int node = blockIdx.x * 8 + wv;
  if (node >= N) return;

  int idx[16];
#pragma unroll
  for (int d = 0; d < 16; ++d) idx[d] = (int)nbr[(size_t)node * 16 + d];

  float meanacc = 0.f;
  for (int h = 0; h < 4; ++h) {
    const float fdv = fd[(size_t)node * 128 + h * 32 + lane];
    const float av = attn[h * 32 + lane];
    float s[16], fcache[16];
#pragma unroll
    for (int d = 0; d < 16; ++d) {
      if (d < 15) __builtin_prefetch(&fs[(size_t)idx[d + 1] * 128 + h * 32 + lane], 0, 0);
      float fsv = fs[(size_t)idx[d] * 128 + h * 32 + lane];
      fcache[d] = fsv;
      float e = fsv + fdv;
      e = e > 0.f ? e : 0.2f * e;        // leaky_relu 0.2
      float p = e * av;
#pragma unroll
      for (int off = 16; off > 0; off >>= 1) p += __shfl_xor(p, off, 32);
      s[d] = p;                           // full dot over O (all lanes hold it)
    }
    float mx = s[0];
#pragma unroll
    for (int d = 1; d < 16; ++d) mx = fmaxf(mx, s[d]);
    float sum = 0.f, a[16];
#pragma unroll
    for (int d = 0; d < 16; ++d) { a[d] = __expf(s[d] - mx); sum += a[d]; }
    const float inv = __builtin_amdgcn_rcpf(sum);
    float acc = 0.f;
#pragma unroll
    for (int d = 0; d < 16; ++d) acc += a[d] * inv * fcache[d];
    acc += bias[h * 32 + lane];
    if (mode == 2) {
      meanacc += acc;
    } else {
      float r = acc > 0.f ? acc : (__expf(acc) - 1.f);   // elu
      out[(size_t)node * 128 + h * 32 + lane] = r;
    }
  }
  if (mode == 2) out[(size_t)node * 32 + lane] = meanacc * 0.25f;
}

// ---------------------------------------------------------------------------
// Final MLP head: [N,32]@[32,16] -> lrelu(0.01) -> @[16,1]
// ---------------------------------------------------------------------------
__global__ __launch_bounds__(256)
void fc_kernel(const float* __restrict__ hin, const float* __restrict__ w1,
               const float* __restrict__ b1, const float* __restrict__ w2,
               const float* __restrict__ b2, float* __restrict__ out, int N) {
  const int n = blockIdx.x * 256 + threadIdx.x;
  if (n >= N) return;
  float v[32];
#pragma unroll
  for (int k = 0; k < 32; ++k) v[k] = hin[(size_t)n * 32 + k];
  float acc = b2[0];
#pragma unroll
  for (int j = 0; j < 16; ++j) {
    float t = b1[j];
#pragma unroll
    for (int k = 0; k < 32; ++k) t += v[k] * w1[k * 16 + j];
    t = t > 0.f ? t : 0.01f * t;
    acc += t * w2[j];
  }
  out[n] = acc;
}

// ---------------------------------------------------------------------------
extern "C" void kernel_launch(void* const* d_in, const int* in_sizes, int n_in,
                              void* d_out, int out_size, void* d_ws, size_t ws_size,
                              hipStream_t stream) {
  (void)n_in; (void)out_size; (void)ws_size;
  const int N = in_sizes[0] / 16;

  const float*     x   = (const float*)d_in[0];
  const long long* nbr = (const long long*)d_in[1];
  const float *l1wih = (const float*)d_in[2],  *l1whh = (const float*)d_in[3];
  const float *l1bih = (const float*)d_in[4],  *l1bhh = (const float*)d_in[5];
  const float *s1ws  = (const float*)d_in[6],  *s1wn  = (const float*)d_in[7],  *s1b = (const float*)d_in[8];
  const float *l2wih = (const float*)d_in[9],  *l2whh = (const float*)d_in[10];
  const float *l2bih = (const float*)d_in[11], *l2bhh = (const float*)d_in[12];
  const float *s2ws  = (const float*)d_in[13], *s2wn  = (const float*)d_in[14], *s2b = (const float*)d_in[15];
  const float *g0s = (const float*)d_in[16], *g0d = (const float*)d_in[17], *g0a = (const float*)d_in[18], *g0b = (const float*)d_in[19];
  const float *g1s = (const float*)d_in[20], *g1d = (const float*)d_in[21], *g1a = (const float*)d_in[22], *g1b = (const float*)d_in[23];
  const float *g2s = (const float*)d_in[24], *g2d = (const float*)d_in[25], *g2a = (const float*)d_in[26], *g2b = (const float*)d_in[27];
  const float *f1w = (const float*)d_in[28], *f1b = (const float*)d_in[29];
  const float *f2w = (const float*)d_in[30], *f2b = (const float*)d_in[31];

  float* ws = (float*)d_ws;
  float* h1 = ws;                          // N*32  (also reused for gat2 output)
  float* h2 = h1 + (size_t)N * 32;         // N*32
  float* fs = h2 + (size_t)N * 32;         // N*128
  float* fd = fs + (size_t)N * 128;        // N*128
  float* gb = fd + (size_t)N * 128;        // N*128

  const int wavesTotal = (N + 15) / 16;            // 3125
  const int sageBlocks = (wavesTotal + 3) / 4;     // 782
  const int edgeBlocks = (N + 7) / 8;              // 6250
  const int fcBlocks   = (N + 255) / 256;

  sage1_kernel<<<sageBlocks, 128, 0, stream>>>(x, nbr, l1wih, l1whh, l1bih, l1bhh, s1ws, s1wn, s1b, h1, N);
  sage2_kernel<<<sageBlocks, 128, 0, stream>>>(h1, nbr, l2wih, l2whh, l2bih, l2bhh, s2ws, s2wn, s2b, h2, N);

  proj_kernel<<<sageBlocks, 128, 0, stream>>>(h2, g0s, fs, N, 32);
  proj_kernel<<<sageBlocks, 128, 0, stream>>>(h2, g0d, fd, N, 32);
  gat_edge_kernel<<<edgeBlocks, 256, 0, stream>>>(fs, fd, nbr, g0a, g0b, gb, 0, N);

  proj_kernel<<<sageBlocks, 128, 0, stream>>>(gb, g1s, fs, N, 128);
  proj_kernel<<<sageBlocks, 128, 0, stream>>>(gb, g1d, fd, N, 128);
  gat_edge_kernel<<<edgeBlocks, 256, 0, stream>>>(fs, fd, nbr, g1a, g1b, gb, 1, N);

  proj_kernel<<<sageBlocks, 128, 0, stream>>>(gb, g2s, fs, N, 128);
  proj_kernel<<<sageBlocks, 128, 0, stream>>>(gb, g2d, fd, N, 128);
  gat_edge_kernel<<<edgeBlocks, 256, 0, stream>>>(fs, fd, nbr, g2a, g2b, h1, 2, N);

  fc_kernel<<<fcBlocks, 256, 0, stream>>>(h1, f1w, f1b, f2w, f2b, (float*)d_out, N);
}
